// AdvancedDynamicFBNETGEN_53420803228011
// MI455X (gfx1250) — matrix operand order repaired and checked by reading, not compiled, not run on hardware
//
#include <hip/hip_runtime.h>
#include <math.h>

// ---------------------------------------------------------------------------
// Model dims (from reference)
// ---------------------------------------------------------------------------
#define TT   16
#define RR   200
#define WW   50
#define FCWT 32
#define EE   128
#define HH   256
#define HGG  256
#define NCC  2
#define KTOP 20
#define NN   (TT * RR)          // 3200

// conv1: (N,1,32,50) -> (N,16,32,50) -> pool -> (N,16,16,25)
// conv2: (N,16,16,25) -> (N,32,16,25) -> pool -> (N,32,8,12)
#define CNT1 (NN * 32 * 50)     // 5,120,000 per-channel BN count, conv1
#define CNT2 (NN * 16 * 25)     // 1,280,000 per-channel BN count, conv2

// ---------------------------------------------------------------------------
// Workspace layout (floats). Total ~34.6M floats (~138 MB); L2-resident.
// ---------------------------------------------------------------------------
#define OFF_SCAL   0L                               // N*32*50      = 5,120,000
#define OFF_H1     (OFF_SCAL   + 5120000L)          // N*16*16*25   = 20,480,000
#define OFF_H2     (OFF_H1     + 20480000L)         // N*32*8*12    = 9,830,400 (2,457,600 reserved)
#define OFF_POOLED (OFF_H2     + 2457600L)          // N*32         = 102,400
#define OFF_EMB    (OFF_POOLED + 102400L)           // N*128        = 409,600
#define OFF_Q      (OFF_EMB    + 409600L)           // 409,600
#define OFF_K      (OFF_Q      + 409600L)           // 409,600
#define OFF_ATTN   (OFF_K      + 409600L)           // 16*200*200   = 640,000
#define OFF_ADJ    (OFF_ATTN   + 640000L)           // 640,000
#define OFF_AN     (OFF_ADJ    + 640000L)           // 640,000
#define OFF_GX1    (OFF_AN     + 640000L)           // N*256        = 819,200
#define OFF_G1     (OFF_GX1    + 819200L)           // 819,200
#define OFF_GX2    (OFF_G1     + 819200L)           // 819,200
#define OFF_G2     (OFF_GX2    + 819200L)           // 819,200
#define OFF_GEMB   (OFF_G2     + 819200L)           // 16*256       = 4,096
#define OFF_STATS  (OFF_GEMB   + 4096L)             // 96: s1[16] q1[16] s2[32] q2[32]

// ---------------------------------------------------------------------------
// WMMA types
// ---------------------------------------------------------------------------
typedef __attribute__((ext_vector_type(16))) _Float16 v16h;
typedef __attribute__((ext_vector_type(8)))  float    v8f;

// ---------------------------------------------------------------------------
// init: zero BN stat accumulators and edge_var output cell
// ---------------------------------------------------------------------------
__global__ void init_kernel(float* stats, float* out) {
    int i = threadIdx.x;
    if (i < 96) stats[i] = 0.0f;
    if (i == 96) out[40002] = 0.0f;
}

// ---------------------------------------------------------------------------
// CWT: per sample n, conv1d with 32 real + 32 imag wavelets (len 50, pad 24/25),
// magnitude -> scal[n][f][p]   (N, 32, 50)
// ---------------------------------------------------------------------------
__global__ __launch_bounds__(256) void cwt_kernel(const float* __restrict__ x,
                                                  const float* __restrict__ wr,
                                                  const float* __restrict__ wi,
                                                  float* __restrict__ scal) {
    const int n = blockIdx.x;
    __shared__ float sx[WW];
    __shared__ float swr[FCWT * WW];
    __shared__ float swi[FCWT * WW];
    for (int i = threadIdx.x; i < WW; i += blockDim.x) sx[i] = x[(long)n * WW + i];
    for (int i = threadIdx.x; i < FCWT * WW; i += blockDim.x) {
        swr[i] = wr[i];
        swi[i] = wi[i];
    }
    __syncthreads();
    for (int idx = threadIdx.x; idx < FCWT * WW; idx += blockDim.x) {
        const int f = idx / WW, p = idx % WW;
        float re = 0.0f, im = 0.0f;
        for (int k = 0; k < WW; ++k) {
            const int q = p + k - 24;                 // pl = 24
            if (q >= 0 && q < WW) {
                const float xv = sx[q];
                re += xv * swr[f * WW + k];
                im += xv * swi[f * WW + k];
            }
        }
        scal[(long)n * FCWT * WW + idx] = sqrtf(re * re + im * im);
    }
}

// ---------------------------------------------------------------------------
// conv1 BN stats: recompute conv (9 MAC) and accumulate per-channel sum/sumsq
// ---------------------------------------------------------------------------
__global__ __launch_bounds__(256) void conv1_stats_kernel(const float* __restrict__ scal,
                                                          const float* __restrict__ w,
                                                          const float* __restrict__ b,
                                                          float* __restrict__ sum1,
                                                          float* __restrict__ sq1) {
    const int n = blockIdx.x;
    __shared__ float img[32 * 50];
    __shared__ float ssum[16], ssq[16];
    for (int i = threadIdx.x; i < 16; i += blockDim.x) { ssum[i] = 0.0f; ssq[i] = 0.0f; }
    for (int i = threadIdx.x; i < 1600; i += blockDim.x) img[i] = scal[(long)n * 1600 + i];
    __syncthreads();
    for (int c = 0; c < 16; ++c) {
        float ps = 0.0f, pq = 0.0f;
        for (int idx = threadIdx.x; idx < 1600; idx += blockDim.x) {
            const int i = idx / 50, j = idx % 50;
            float v = b[c];
            for (int di = 0; di < 3; ++di)
                for (int dj = 0; dj < 3; ++dj) {
                    const int ii = i + di - 1, jj = j + dj - 1;
                    if (ii >= 0 && ii < 32 && jj >= 0 && jj < 50)
                        v += img[ii * 50 + jj] * w[c * 9 + di * 3 + dj];
                }
            ps += v;
            pq += v * v;
        }
        atomicAdd(&ssum[c], ps);
        atomicAdd(&ssq[c], pq);
    }
    __syncthreads();
    for (int c = threadIdx.x; c < 16; c += blockDim.x) {
        atomicAdd(&sum1[c], ssum[c]);
        atomicAdd(&sq1[c], ssq[c]);
    }
}

// ---------------------------------------------------------------------------
// conv1 normalize + relu + 2x2 maxpool -> h1 (N,16,16,25)
// ---------------------------------------------------------------------------
__global__ __launch_bounds__(256) void conv1_npool_kernel(const float* __restrict__ scal,
                                                          const float* __restrict__ w,
                                                          const float* __restrict__ b,
                                                          const float* __restrict__ sum1,
                                                          const float* __restrict__ sq1,
                                                          const float* __restrict__ g,
                                                          const float* __restrict__ beta,
                                                          float* __restrict__ h1) {
    const long idx = (long)blockIdx.x * blockDim.x + threadIdx.x;
    if (idx >= (long)NN * 16 * 16 * 25) return;
    const int j = (int)(idx % 25);
    const int i = (int)((idx / 25) % 16);
    const int c = (int)((idx / 400) % 16);
    const int n = (int)(idx / 6400);
    const float mu = sum1[c] / (float)CNT1;
    const float var = sq1[c] / (float)CNT1 - mu * mu;
    const float is = rsqrtf(var + 1e-5f) * g[c];
    const float bt = beta[c];
    float mx = 0.0f;   // post-relu values are >= 0
    for (int pi = 0; pi < 2; ++pi)
        for (int pj = 0; pj < 2; ++pj) {
            const int ii0 = 2 * i + pi, jj0 = 2 * j + pj;
            float v = b[c];
            for (int di = 0; di < 3; ++di)
                for (int dj = 0; dj < 3; ++dj) {
                    const int ii = ii0 + di - 1, jj = jj0 + dj - 1;
                    if (ii >= 0 && ii < 32 && jj >= 0 && jj < 50)
                        v += scal[(long)n * 1600 + ii * 50 + jj] * w[c * 9 + di * 3 + dj];
                }
            v = (v - mu) * is + bt;
            v = v > 0.0f ? v : 0.0f;
            mx = fmaxf(mx, v);
        }
    h1[idx] = mx;
}

// ---------------------------------------------------------------------------
// conv2 BN stats (input h1 (N,16,16,25), weights (32,16,3,3))
// ---------------------------------------------------------------------------
__global__ __launch_bounds__(256) void conv2_stats_kernel(const float* __restrict__ h1,
                                                          const float* __restrict__ w2,
                                                          const float* __restrict__ b2,
                                                          float* __restrict__ sum2,
                                                          float* __restrict__ sq2) {
    const int n = blockIdx.x;
    __shared__ float img[16 * 16 * 25];      // 25.6 KB
    __shared__ float ssum[32], ssq[32];
    for (int i = threadIdx.x; i < 32; i += blockDim.x) { ssum[i] = 0.0f; ssq[i] = 0.0f; }
    for (int i = threadIdx.x; i < 6400; i += blockDim.x) img[i] = h1[(long)n * 6400 + i];
    __syncthreads();
    for (int out = threadIdx.x; out < 32 * 400; out += blockDim.x) {
        const int c = out / 400;
        const int i = (out % 400) / 25;
        const int j = out % 25;
        float v = b2[c];
        for (int ic = 0; ic < 16; ++ic)
            for (int di = 0; di < 3; ++di)
                for (int dj = 0; dj < 3; ++dj) {
                    const int ii = i + di - 1, jj = j + dj - 1;
                    if (ii >= 0 && ii < 16 && jj >= 0 && jj < 25)
                        v += img[ic * 400 + ii * 25 + jj] * w2[c * 144 + ic * 9 + di * 3 + dj];
                }
        atomicAdd(&ssum[c], v);
        atomicAdd(&ssq[c], v * v);
    }
    __syncthreads();
    for (int c = threadIdx.x; c < 32; c += blockDim.x) {
        atomicAdd(&sum2[c], ssum[c]);
        atomicAdd(&sq2[c], ssq[c]);
    }
}

// ---------------------------------------------------------------------------
// conv2 normalize + relu + 2x2 maxpool -> h2 (N,32,8,12)
// ---------------------------------------------------------------------------
__global__ __launch_bounds__(256) void conv2_npool_kernel(const float* __restrict__ h1,
                                                          const float* __restrict__ w2,
                                                          const float* __restrict__ b2,
                                                          const float* __restrict__ sum2,
                                                          const float* __restrict__ sq2,
                                                          const float* __restrict__ g,
                                                          const float* __restrict__ beta,
                                                          float* __restrict__ h2) {
    const long idx = (long)blockIdx.x * blockDim.x + threadIdx.x;
    if (idx >= (long)NN * 32 * 8 * 12) return;
    const int j = (int)(idx % 12);
    const int i = (int)((idx / 12) % 8);
    const int c = (int)((idx / 96) % 32);
    const int n = (int)(idx / 3072);
    const float mu = sum2[c] / (float)CNT2;
    const float var = sq2[c] / (float)CNT2 - mu * mu;
    const float is = rsqrtf(var + 1e-5f) * g[c];
    const float bt = beta[c];
    float mx = 0.0f;
    for (int pi = 0; pi < 2; ++pi)
        for (int pj = 0; pj < 2; ++pj) {
            const int ii0 = 2 * i + pi, jj0 = 2 * j + pj;
            float v = b2[c];
            for (int ic = 0; ic < 16; ++ic)
                for (int di = 0; di < 3; ++di)
                    for (int dj = 0; dj < 3; ++dj) {
                        const int ii = ii0 + di - 1, jj = jj0 + dj - 1;
                        if (ii >= 0 && ii < 16 && jj >= 0 && jj < 25)
                            v += h1[(long)n * 6400 + ic * 400 + ii * 25 + jj] *
                                 w2[c * 144 + ic * 9 + di * 3 + dj];
                    }
            v = (v - mu) * is + bt;
            v = v > 0.0f ? v : 0.0f;
            mx = fmaxf(mx, v);
        }
    h2[idx] = mx;
}

// ---------------------------------------------------------------------------
// global average pool over (8,12) -> pooled (N,32)
// ---------------------------------------------------------------------------
__global__ void pooled_mean_kernel(const float* __restrict__ h2, float* __restrict__ pooled) {
    const int idx = blockIdx.x * blockDim.x + threadIdx.x;
    if (idx >= NN * 32) return;
    const int c = idx % 32, n = idx / 32;
    float s = 0.0f;
    for (int k = 0; k < 96; ++k) s += h2[(long)n * 3072 + c * 96 + k];
    pooled[idx] = s * (1.0f / 96.0f);
}

// ---------------------------------------------------------------------------
// Batched WMMA GEMM, LDS-staged.  One wave per 32x32 C macro-tile (2x2
// register blocking -> 4 v_wmma_f32_16x16x32_f16 per 32-wide K step).
//
// Data path: BRANCHLESS clamped global_load_b128 (addresses clamped in
// bounds -> loads form one clause; OOB lanes zeroed with cndmask selects)
// -> LDS (rows padded to 33 floats, conflict-free) -> branch-free fragment
// gather + packed cvt -> WMMA (f32 accumulate).  Fragment layouts per ISA:
//   A frag : lane -> M = (lane&15); elem j -> K = (j&7) + (hi<<3) + ((j>>3)<<4)
//   B frag : lane -> N = (lane&15); elem j -> K = (j&7) + (hi<<3) + ((j>>3)<<4)
//   C frag : elem r -> M = r + 8*hi ; N = (lane&15)
// Requires K % 4 == 0 and N % 4 == 0 and 16B-aligned rows (true for every
// GEMM in this net: K in {32,128,200,256}, N in {128,200,256}).
// act: 0 none, 1 relu, 2 leaky-relu(0.2)
// ---------------------------------------------------------------------------
__global__ __launch_bounds__(32) void wmma_gemm_kernel(
        const float* __restrict__ A, long batchA, int lda,
        const float* __restrict__ B, long batchB, int ldb, int transB,
        float* __restrict__ C, long batchC, int ldc,
        int M, int N, int Kd,
        const float* __restrict__ bias, float scale, int act) {
    const int lane = threadIdx.x;
    const int hi = lane >> 4;
    const int lo = lane & 15;
    const int tm = blockIdx.x * 32;
    const int tn = blockIdx.y * 32;
    A += (long)blockIdx.z * batchA;
    B += (long)blockIdx.z * batchB;
    C += (long)blockIdx.z * batchC;

    __shared__ float As[32][33];   // [row][k]
    __shared__ float Bs[32][33];   // [k][col]

    v8f acc[2][2] = {};

    const int arow = tm + lane;                     // A row staged by this lane
    const bool arow_ok = arow < M;
    const int arow_c = arow_ok ? arow : (M - 1);    // clamped (always legal)
    const int bcol = tn + lane;                     // B col staged (transB)
    const bool bcol_ok = bcol < N;
    const int bcol_c = bcol_ok ? bcol : (N - 1);

    for (int k0 = 0; k0 < Kd; k0 += 32) {
        // uniform-guard prefetch of the next K chunk (-> global_prefetch_b8)
        if (k0 + 32 < Kd)
            __builtin_prefetch(&A[(long)arow_c * lda + k0 + 32], 0, 3);

        // ---- stage A tile (32 rows x 32 k): clamped b128 loads, then zero
        //      OOB lanes, then LDS stores (loads clause together, one wait)
        {
            float4 ra[8];
#pragma unroll
            for (int q = 0; q < 8; ++q) {
                const int k = k0 + 4 * q;
                const int kc = k < Kd ? k : (Kd - 4);
                ra[q] = *reinterpret_cast<const float4*>(&A[(long)arow_c * lda + kc]);
            }
#pragma unroll
            for (int q = 0; q < 8; ++q) {
                const bool ok = arow_ok && (k0 + 4 * q) < Kd;
                As[lane][4 * q + 0] = ok ? ra[q].x : 0.0f;
                As[lane][4 * q + 1] = ok ? ra[q].y : 0.0f;
                As[lane][4 * q + 2] = ok ? ra[q].z : 0.0f;
                As[lane][4 * q + 3] = ok ? ra[q].w : 0.0f;
            }
        }
        // ---- stage B tile (32 k x 32 cols), same branchless scheme
        if (transB) {
            float4 rb[8];
#pragma unroll
            for (int q = 0; q < 8; ++q) {
                const int k = k0 + 4 * q;
                const int kc = k < Kd ? k : (Kd - 4);
                rb[q] = *reinterpret_cast<const float4*>(&B[(long)bcol_c * ldb + kc]);
            }
#pragma unroll
            for (int q = 0; q < 8; ++q) {
                const bool ok = bcol_ok && (k0 + 4 * q) < Kd;
                Bs[4 * q + 0][lane] = ok ? rb[q].x : 0.0f;
                Bs[4 * q + 1][lane] = ok ? rb[q].y : 0.0f;
                Bs[4 * q + 2][lane] = ok ? rb[q].z : 0.0f;
                Bs[4 * q + 3][lane] = ok ? rb[q].w : 0.0f;
            }
        } else {
            const int krow = k0 + lane;
            const bool k_ok = krow < Kd;
            const int krow_c = k_ok ? krow : (Kd - 1);
            float4 rb[8];
#pragma unroll
            for (int q = 0; q < 8; ++q) {
                const int c = tn + 4 * q;
                const int cc = c < N ? c : (N - 4);
                rb[q] = *reinterpret_cast<const float4*>(&B[(long)krow_c * ldb + cc]);
            }
#pragma unroll
            for (int q = 0; q < 8; ++q) {
                const bool ok = k_ok && (tn + 4 * q) < N;
                Bs[lane][4 * q + 0] = ok ? rb[q].x : 0.0f;
                Bs[lane][4 * q + 1] = ok ? rb[q].y : 0.0f;
                Bs[lane][4 * q + 2] = ok ? rb[q].z : 0.0f;
                Bs[lane][4 * q + 3] = ok ? rb[q].w : 0.0f;
            }
        }
        __syncthreads();

        // ---- branch-free fragment build + 2x2 WMMA
        v16h a[2], b[2];
#pragma unroll
        for (int mi = 0; mi < 2; ++mi)
#pragma unroll
            for (int j = 0; j < 16; ++j) {
                const int ka = (j & 7) + (hi << 3) + ((j >> 3) << 4);
                a[mi][j] = (_Float16)As[lo + 16 * mi][ka];
            }
#pragma unroll
        for (int ni = 0; ni < 2; ++ni)
#pragma unroll
            for (int j = 0; j < 16; ++j) {
                const int kb = (j & 7) + (hi << 3) + ((j >> 3) << 4);
                b[ni][j] = (_Float16)Bs[kb][lo + 16 * ni];
            }
        acc[0][0] = __builtin_amdgcn_wmma_f32_16x16x32_f16(false, a[0], false, b[0],
                                                           (short)0, acc[0][0], false, false);
        acc[0][1] = __builtin_amdgcn_wmma_f32_16x16x32_f16(false, a[0], false, b[1],
                                                           (short)0, acc[0][1], false, false);
        acc[1][0] = __builtin_amdgcn_wmma_f32_16x16x32_f16(false, a[1], false, b[0],
                                                           (short)0, acc[1][0], false, false);
        acc[1][1] = __builtin_amdgcn_wmma_f32_16x16x32_f16(false, a[1], false, b[1],
                                                           (short)0, acc[1][1], false, false);
        __syncthreads();
    }

    // ---- epilogue: scale, bias, activation, guarded store
#pragma unroll
    for (int mi = 0; mi < 2; ++mi)
#pragma unroll
        for (int ni = 0; ni < 2; ++ni)
#pragma unroll
            for (int r = 0; r < 8; ++r) {
                const int m = tm + 16 * mi + r + (hi << 3);
                const int n = tn + 16 * ni + lo;
                if (m < M && n < N) {
                    float v = acc[mi][ni][r] * scale;
                    if (bias) v += bias[n];
                    if (act == 1) v = v > 0.0f ? v : 0.0f;
                    else if (act == 2) v = v > 0.0f ? v : 0.2f * v;
                    C[(long)m * ldc + n] = v;
                }
            }
}

// ---------------------------------------------------------------------------
// top-k (K=20, lax.top_k tie-break: value desc, index asc) mask + softmax
// one block per (t,r) row of attn (length 200)
// ---------------------------------------------------------------------------
__global__ __launch_bounds__(256) void topk_softmax_kernel(const float* __restrict__ attn,
                                                           float* __restrict__ adj) {
    const long row = blockIdx.x;                  // t*200 + r
    const float* a = attn + row * RR;
    __shared__ float vals[RR];
    __shared__ unsigned char sel[RR];
    __shared__ float redv[256];
    __shared__ int   redi[256];
    const int tid = threadIdx.x;
    for (int i = tid; i < RR; i += 256) { vals[i] = a[i]; sel[i] = 0; }
    __syncthreads();

    for (int it = 0; it < KTOP; ++it) {
        float bv = -INFINITY;
        int bi = RR;
        for (int i = tid; i < RR; i += 256)
            if (!sel[i]) {
                const float v = vals[i];
                if (v > bv || (v == bv && i < bi)) { bv = v; bi = i; }
            }
        redv[tid] = bv; redi[tid] = bi;
        __syncthreads();
        for (int off = 128; off > 0; off >>= 1) {
            if (tid < off) {
                const float v2 = redv[tid + off];
                const int   i2 = redi[tid + off];
                if (v2 > redv[tid] || (v2 == redv[tid] && i2 < redi[tid])) {
                    redv[tid] = v2; redi[tid] = i2;
                }
            }
            __syncthreads();
        }
        if (tid == 0 && redi[0] < RR) sel[redi[0]] = 1;
        __syncthreads();
    }

    // masked = sel && v!=0 ? v : -1e9 ; then softmax
    float lm = -INFINITY;
    for (int i = tid; i < RR; i += 256) {
        const float mv = (sel[i] && vals[i] != 0.0f) ? vals[i] : -1e9f;
        vals[i] = mv;
        lm = fmaxf(lm, mv);
    }
    redv[tid] = lm;
    __syncthreads();
    for (int off = 128; off > 0; off >>= 1) {
        if (tid < off) redv[tid] = fmaxf(redv[tid], redv[tid + off]);
        __syncthreads();
    }
    const float mx = redv[0];
    __syncthreads();
    float ls = 0.0f;
    for (int i = tid; i < RR; i += 256) {
        const float e = expf(vals[i] - mx);
        vals[i] = e;
        ls += e;
    }
    redv[tid] = ls;
    __syncthreads();
    for (int off = 128; off > 0; off >>= 1) {
        if (tid < off) redv[tid] += redv[tid + off];
        __syncthreads();
    }
    const float inv = 1.0f / redv[0];
    for (int i = tid; i < RR; i += 256) adj[row * RR + i] = vals[i] * inv;
}

// ---------------------------------------------------------------------------
// self-loop fix + transpose + symmetric degree normalization:
// An[t][r][s] = dinv[r] * adj_sl[t][s][r] * dinv[s]
// ---------------------------------------------------------------------------
__global__ __launch_bounds__(256) void adj_norm_kernel(const float* __restrict__ adj,
                                                       float* __restrict__ An) {
    const int t = blockIdx.x;
    const float* a = adj + (long)t * RR * RR;
    __shared__ float dinv[RR];
    for (int r = threadIdx.x; r < RR; r += blockDim.x) {
        float d = 0.0f;
        for (int s = 0; s < RR; ++s) {
            float v = a[s * RR + r];
            if (s == r && v == 0.0f) v = 1.0f;
            d += v;
        }
        const float dd = d > 0.0f ? d : 1.0f;
        dinv[r] = rsqrtf(dd);
    }
    __syncthreads();
    for (int idx = threadIdx.x; idx < RR * RR; idx += blockDim.x) {
        const int r = idx / RR, s = idx % RR;
        float v = a[s * RR + r];                // Bm[r][s] = adj_sl[s][r]
        if (s == r && v == 0.0f) v = 1.0f;
        An[(long)t * RR * RR + idx] = dinv[r] * v * dinv[s];
    }
}

// ---------------------------------------------------------------------------
// g_emb[t] = mean over r of g2[t*200+r][:]   -> (16,256)
// ---------------------------------------------------------------------------
__global__ void gemb_mean_kernel(const float* __restrict__ g2, float* __restrict__ gemb) {
    const int t = blockIdx.x;
    const int c = threadIdx.x;                  // 256 threads
    float s = 0.0f;
    for (int r = 0; r < RR; ++r) s += g2[((long)t * RR + r) * HH + c];
    gemb[t * HH + c] = s * (1.0f / (float)RR);
}

// ---------------------------------------------------------------------------
// GRU over T=16 steps (H=HG=256) + classifier -> d_out[0..1]
// single block, 256 threads, h kept in LDS
// ---------------------------------------------------------------------------
__global__ __launch_bounds__(256) void gru_kernel(const float* __restrict__ gemb,
                                                  const float* __restrict__ wih,
                                                  const float* __restrict__ whh,
                                                  const float* __restrict__ bih,
                                                  const float* __restrict__ bhh,
                                                  const float* __restrict__ cls_w,
                                                  const float* __restrict__ cls_b,
                                                  float* __restrict__ out) {
    __shared__ float h[HGG], xt[HH], gi[3 * HGG], gh[3 * HGG];
    const int tid = threadIdx.x;
    h[tid] = 0.0f;
    __syncthreads();
    for (int t = 0; t < TT; ++t) {
        xt[tid] = gemb[t * HH + tid];
        __syncthreads();
        for (int oo = 0; oo < 3; ++oo) {
            const int o = oo * HGG + tid;
            float si = bih[o], sh = bhh[o];
            for (int j = 0; j < HH; ++j) si += wih[(long)o * HH + j] * xt[j];
            for (int j = 0; j < HGG; ++j) sh += whh[(long)o * HGG + j] * h[j];
            gi[o] = si;
            gh[o] = sh;
        }
        __syncthreads();
        const float r = 1.0f / (1.0f + expf(-(gi[tid] + gh[tid])));
        const float z = 1.0f / (1.0f + expf(-(gi[HGG + tid] + gh[HGG + tid])));
        const float nn = tanhf(gi[2 * HGG + tid] + r * gh[2 * HGG + tid]);
        const float hn = (1.0f - z) * nn + z * h[tid];
        __syncthreads();
        h[tid] = hn;
        __syncthreads();
    }
    if (tid < NCC) {
        float s = cls_b[tid];
        for (int j = 0; j < HGG; ++j) s += h[j] * cls_w[tid * HGG + j];
        out[tid] = s;
    }
}

// ---------------------------------------------------------------------------
// avg_adj (mean over t) -> out[2..40001]; edge_var (mean of ddof=1 var over t)
// atomically accumulated into out[40002]
// ---------------------------------------------------------------------------
__global__ __launch_bounds__(256) void avgvar_kernel(const float* __restrict__ adj,
                                                     float* __restrict__ out) {
    const int idx = blockIdx.x * 256 + threadIdx.x;      // 0..39999
    __shared__ float part[256];
    float vloc = 0.0f;
    if (idx < RR * RR) {
        float m = 0.0f;
        for (int t = 0; t < TT; ++t) m += adj[(long)t * RR * RR + idx];
        m *= (1.0f / (float)TT);
        float v = 0.0f;
        for (int t = 0; t < TT; ++t) {
            const float d = adj[(long)t * RR * RR + idx] - m;
            v += d * d;
        }
        v *= (1.0f / (float)(TT - 1));                   // ddof = 1
        out[2 + idx] = m;
        vloc = v * (1.0f / (float)(RR * RR));
    }
    part[threadIdx.x] = vloc;
    __syncthreads();
    for (int off = 128; off > 0; off >>= 1) {
        if (threadIdx.x < off) part[threadIdx.x] += part[threadIdx.x + off];
        __syncthreads();
    }
    if (threadIdx.x == 0) atomicAdd(&out[40002], part[0]);
}

// ---------------------------------------------------------------------------
// launch
// ---------------------------------------------------------------------------
extern "C" void kernel_launch(void* const* d_in, const int* in_sizes, int n_in,
                              void* d_out, int out_size, void* d_ws, size_t ws_size,
                              hipStream_t stream) {
    (void)in_sizes; (void)n_in; (void)out_size; (void)ws_size;
    const float* time_windows = (const float*)d_in[0];
    const float* wav_real = (const float*)d_in[1];
    const float* wav_imag = (const float*)d_in[2];
    const float* conv1_w = (const float*)d_in[3];
    const float* conv1_b = (const float*)d_in[4];
    const float* bn1_g   = (const float*)d_in[5];
    const float* bn1_b   = (const float*)d_in[6];
    const float* conv2_w = (const float*)d_in[7];
    const float* conv2_b = (const float*)d_in[8];
    const float* bn2_g   = (const float*)d_in[9];
    const float* bn2_b   = (const float*)d_in[10];
    const float* fc_w    = (const float*)d_in[11];
    const float* fc_b    = (const float*)d_in[12];
    const float* q_w     = (const float*)d_in[13];
    const float* q_b     = (const float*)d_in[14];
    const float* k_w     = (const float*)d_in[15];
    const float* k_b     = (const float*)d_in[16];
    const float* gcn1_w  = (const float*)d_in[17];
    const float* gcn1_b  = (const float*)d_in[18];
    const float* gcn2_w  = (const float*)d_in[19];
    const float* gcn2_b  = (const float*)d_in[20];
    const float* gru_wih = (const float*)d_in[21];
    const float* gru_whh = (const float*)d_in[22];
    const float* gru_bih = (const float*)d_in[23];
    const float* gru_bhh = (const float*)d_in[24];
    const float* cls_w   = (const float*)d_in[25];
    const float* cls_b   = (const float*)d_in[26];

    float* ws  = (float*)d_ws;
    float* out = (float*)d_out;

    float* scal   = ws + OFF_SCAL;
    float* h1     = ws + OFF_H1;
    float* h2     = ws + OFF_H2;
    float* pooled = ws + OFF_POOLED;
    float* emb    = ws + OFF_EMB;
    float* Q      = ws + OFF_Q;
    float* Kp     = ws + OFF_K;
    float* attn   = ws + OFF_ATTN;
    float* adj    = ws + OFF_ADJ;
    float* An     = ws + OFF_AN;
    float* gx1    = ws + OFF_GX1;
    float* g1     = ws + OFF_G1;
    float* gx2    = ws + OFF_GX2;
    float* g2     = ws + OFF_G2;
    float* gemb   = ws + OFF_GEMB;
    float* stats  = ws + OFF_STATS;
    float* sum1 = stats +  0;
    float* sq1  = stats + 16;
    float* sum2 = stats + 32;
    float* sq2  = stats + 64;

    init_kernel<<<1, 128, 0, stream>>>(stats, out);

    cwt_kernel<<<NN, 256, 0, stream>>>(time_windows, wav_real, wav_imag, scal);

    conv1_stats_kernel<<<NN, 256, 0, stream>>>(scal, conv1_w, conv1_b, sum1, sq1);
    {
        const long tot = (long)NN * 16 * 16 * 25;
        conv1_npool_kernel<<<(int)((tot + 255) / 256), 256, 0, stream>>>(
            scal, conv1_w, conv1_b, sum1, sq1, bn1_g, bn1_b, h1);
    }
    conv2_stats_kernel<<<NN, 256, 0, stream>>>(h1, conv2_w, conv2_b, sum2, sq2);
    {
        const long tot = (long)NN * 32 * 8 * 12;
        conv2_npool_kernel<<<(int)((tot + 255) / 256), 256, 0, stream>>>(
            h1, conv2_w, conv2_b, sum2, sq2, bn2_g, bn2_b, h2);
    }
    pooled_mean_kernel<<<(NN * 32 + 255) / 256, 256, 0, stream>>>(h2, pooled);

    // emb = pooled @ fc_w.T + fc_b        (3200x32)(32x128)
    wmma_gemm_kernel<<<dim3(100, 4, 1), 32, 0, stream>>>(
        pooled, 0, 32, fc_w, 0, 32, 1, emb, 0, EE, NN, EE, 32, fc_b, 1.0f, 0);
    // Q = emb @ q_w.T + q_b ; Kp = emb @ k_w.T + k_b
    wmma_gemm_kernel<<<dim3(100, 4, 1), 32, 0, stream>>>(
        emb, 0, EE, q_w, 0, EE, 1, Q, 0, EE, NN, EE, EE, q_b, 1.0f, 0);
    wmma_gemm_kernel<<<dim3(100, 4, 1), 32, 0, stream>>>(
        emb, 0, EE, k_w, 0, EE, 1, Kp, 0, EE, NN, EE, EE, k_b, 1.0f, 0);
    // attn[t] = leaky_relu( Q_t @ Kp_t^T / sqrt(E), 0.2 )   batched over t
    wmma_gemm_kernel<<<dim3(7, 7, TT), 32, 0, stream>>>(
        Q, (long)RR * EE, EE, Kp, (long)RR * EE, EE, 1,
        attn, (long)RR * RR, RR, RR, RR, EE, nullptr,
        1.0f / sqrtf((float)EE), 2);

    topk_softmax_kernel<<<TT * RR, 256, 0, stream>>>(attn, adj);
    adj_norm_kernel<<<TT, 256, 0, stream>>>(adj, An);

    // gx1 = emb @ gcn1_w                  (3200x128)(128x256)
    wmma_gemm_kernel<<<dim3(100, 8, 1), 32, 0, stream>>>(
        emb, 0, EE, gcn1_w, 0, HH, 0, gx1, 0, HH, NN, HH, EE, nullptr, 1.0f, 0);
    // g1[t] = relu(An_t @ gx1_t + gcn1_b)  batched (200x200)(200x256)
    wmma_gemm_kernel<<<dim3(7, 8, TT), 32, 0, stream>>>(
        An, (long)RR * RR, RR, gx1, (long)RR * HH, HH, 0,
        g1, (long)RR * HH, HH, RR, HH, RR, gcn1_b, 1.0f, 1);
    // gx2 = g1 @ gcn2_w                   (3200x256)(256x256)
    wmma_gemm_kernel<<<dim3(100, 8, 1), 32, 0, stream>>>(
        g1, 0, HH, gcn2_w, 0, HH, 0, gx2, 0, HH, NN, HH, HH, nullptr, 1.0f, 0);
    // g2[t] = relu(An_t @ gx2_t + gcn2_b)
    wmma_gemm_kernel<<<dim3(7, 8, TT), 32, 0, stream>>>(
        An, (long)RR * RR, RR, gx2, (long)RR * HH, HH, 0,
        g2, (long)RR * HH, HH, RR, HH, RR, gcn2_b, 1.0f, 1);

    gemb_mean_kernel<<<TT, HH, 0, stream>>>(g2, gemb);
    gru_kernel<<<1, 256, 0, stream>>>(gemb, gru_wih, gru_whh, gru_bih, gru_bhh,
                                      cls_w, cls_b, out);
    avgvar_kernel<<<(RR * RR + 255) / 256, 256, 0, stream>>>(adj, out);
}
